// SelfAttention_40269613367318
// MI455X (gfx1250) — matrix operand order repaired and checked by reading, not compiled, not run on hardware
//
#include <hip/hip_runtime.h>

// ---------------------------------------------------------------------------
// MI455X (gfx1250) self-attention: bf16 WMMA + TDM tensor loads + async-LDS.
//   1) ln_kernel            : LayerNorm(x) -> xn (bf16)
//   2) cvt_transpose_kernel : weights fp32 [K,N] -> bf16 [N,K]
//   3) gemm_kernel          : double-buffered; A tile via TENSOR_LOAD_TO_LDS,
//                             B tile via GLOBAL_LOAD_ASYNC_TO_LDS_B128
//   4) colstats_kernel      : per-key-column softmax stats over QUERY axis
//   5) transpose_v_kernel   : V -> vt[b][hkv][dv][S] (K-major for B operand)
//   6) attnout_kernel       : barrier-free P relayout + PV WMMA
//   7) gemm_kernel<f32>     : attn @ Wo + bo -> d_out (fp32)
// ---------------------------------------------------------------------------

typedef __bf16 bf16;
typedef __attribute__((ext_vector_type(16))) __bf16 v16bf;
typedef __attribute__((ext_vector_type(8)))  __bf16 v8bf;
typedef __attribute__((ext_vector_type(8)))  float  v8f;
typedef __attribute__((ext_vector_type(4)))  unsigned su4;
typedef __attribute__((ext_vector_type(8)))  unsigned su8;

constexpr int BB  = 2;
constexpr int S   = 2048;
constexpr int D   = 2048;
constexpr int HQ  = 32;
constexpr int HKV = 8;
constexpr int HD  = 64;
constexpr int HDV = 64;
constexpr float EPS    = 1e-5f;
constexpr float QSCALE = 0.125f; // 1/sqrt(HD)
constexpr float NEGBIG = -1e30f;

// ---------------------------------------------------------------------------
// CDNA5 async global->LDS copy (16B/lane), tracked by ASYNCcnt.
// Generic LDS pointers carry the LDS byte offset in their low 32 bits.
// ---------------------------------------------------------------------------
__device__ __forceinline__ void async_copy16(void* lds, const void* g) {
  unsigned loff = (unsigned)(unsigned long long)lds;
  unsigned long long ga = (unsigned long long)g;
  asm volatile("global_load_async_to_lds_b128 %0, %1, off"
               :: "v"(loff), "v"(ga) : "memory");
}
template <int N>
__device__ __forceinline__ void async_wait() {
  asm volatile("s_wait_asynccnt %0" :: "n"(N) : "memory");
}

// ---------------------------------------------------------------------------
// CDNA5 Tensor Data Mover: load a 2D tile (32 elems/row x 128 rows, bf16)
// from global (row stride = rowstride elems) into LDS with hardware row
// padding of 4 DWORDs every 16 DWORDs (=> 40-element LDS rows).
// Descriptor per CDNA5 ISA. Wave-level op.
// ---------------------------------------------------------------------------
__device__ __forceinline__ void tdm_load_tile_128x32(void* lds, const void* g,
                                                     unsigned rowstride) {
  unsigned long long ga = (unsigned long long)g;
  su4 g0;
  g0[0] = 1u;                                   // count=1 (valid descriptor)
  g0[1] = (unsigned)(unsigned long long)lds;    // lds_addr (bytes)
  g0[2] = (unsigned)ga;                         // global_addr[31:0]
  g0[3] = (unsigned)(ga >> 32) | (2u << 30);    // global_addr[56:32], type=2
  su8 g1;
  g1[0] = (1u << 16)        // data_size = 1 (2 bytes)
        | (1u << 20)        // pad_enable
        | (3u << 22)        // pad_interval: 16 DWORDs (64B) of data...
        | (3u << 25);       // pad_amount:   4 DWORDs (16B) of pad
  g1[1] = (32u << 16);      // tensor_dim0 = 32 (lo16)
  g1[2] = (128u << 16);     // tensor_dim0 hi = 0 | tensor_dim1 = 128 (lo16)
  g1[3] = (32u << 16);      // tensor_dim1 hi = 0 | tile_dim0 = 32
  g1[4] = 128u;             // tile_dim1 = 128 | tile_dim2 = 0
  g1[5] = rowstride;        // tensor_dim0_stride[31:0] (elements)
  g1[6] = 0u;
  g1[7] = 0u;
  su4 gz;
  gz[0] = 0u; gz[1] = 0u; gz[2] = 0u; gz[3] = 0u;
  asm volatile("tensor_load_to_lds %0, %1, %2, %3"
               :: "s"(g0), "s"(g1), "s"(gz), "s"(gz) : "memory");
}
template <int N>
__device__ __forceinline__ void tensor_wait() {
  __builtin_amdgcn_s_wait_tensorcnt((short)N);
}

// ---------------------------------------------------------------------------
// WMMA helpers (CDNA5 16x16x32 bf16, f32 accumulate)
// ---------------------------------------------------------------------------
__device__ __forceinline__ v8f wmma_bf16(v16bf a, v16bf b, v8f c) {
  return __builtin_amdgcn_wmma_f32_16x16x32_bf16(false, a, false, b, (short)0, c,
                                                 false, false);
}

__device__ __forceinline__ v16bf load_frag_a16x32(const bf16* tile, int ldm) {
  const int lane = threadIdx.x & 31;
  const int row  = lane & 15;
  const int kb   = (lane < 16) ? 0 : 16;
  const bf16* p  = tile + row * ldm + kb;
  v8bf lo = *(const v8bf*)(p);
  v8bf hi = *(const v8bf*)(p + 8);
  v16bf r;
#pragma unroll
  for (int i = 0; i < 8; ++i) { r[i] = lo[i]; r[i + 8] = hi[i]; }
  return r;
}

__device__ __forceinline__ v16bf load_frag_b32x16(const bf16* tileT, int ldn) {
  const int lane = threadIdx.x & 31;
  const int n    = lane & 15;
  const int kb   = (lane < 16) ? 0 : 16;
  const bf16* p  = tileT + n * ldn + kb;
  v8bf lo = *(const v8bf*)(p);
  v8bf hi = *(const v8bf*)(p + 8);
  v16bf r;
#pragma unroll
  for (int i = 0; i < 8; ++i) { r[i] = lo[i]; r[i + 8] = hi[i]; }
  return r;
}

// ---------------------------------------------------------------------------
// 1) LayerNorm
// ---------------------------------------------------------------------------
__global__ void ln_kernel(const float* __restrict__ x,
                          const float* __restrict__ g,
                          const float* __restrict__ bta,
                          bf16* __restrict__ xn) {
  const int row = blockIdx.x;
  const int tid = threadIdx.x;
  const float* xr = x + (size_t)row * D;

  __shared__ float rbuf[256];

  float lsum = 0.f, lsq = 0.f;
  const float4* x4 = (const float4*)xr;
#pragma unroll
  for (int c = 0; c < 2; ++c) {
    float4 v = x4[tid + c * 256];
    lsum += v.x + v.y + v.z + v.w;
    lsq  += v.x * v.x + v.y * v.y + v.z * v.z + v.w * v.w;
  }
  rbuf[tid] = lsum; __syncthreads();
  for (int off = 128; off > 0; off >>= 1) {
    if (tid < off) rbuf[tid] += rbuf[tid + off];
    __syncthreads();
  }
  const float mean = rbuf[0] * (1.0f / D); __syncthreads();
  rbuf[tid] = lsq; __syncthreads();
  for (int off = 128; off > 0; off >>= 1) {
    if (tid < off) rbuf[tid] += rbuf[tid + off];
    __syncthreads();
  }
  const float var  = rbuf[0] * (1.0f / D) - mean * mean;
  const float rstd = __frsqrt_rn(var + EPS);

  bf16* xo = xn + (size_t)row * D;
#pragma unroll
  for (int c = 0; c < 2; ++c) {
    float4 v = x4[tid + c * 256];
    int j = (tid + c * 256) * 4;
    xo[j + 0] = (bf16)((v.x - mean) * rstd * g[j + 0] + bta[j + 0]);
    xo[j + 1] = (bf16)((v.y - mean) * rstd * g[j + 1] + bta[j + 1]);
    xo[j + 2] = (bf16)((v.z - mean) * rstd * g[j + 2] + bta[j + 2]);
    xo[j + 3] = (bf16)((v.w - mean) * rstd * g[j + 3] + bta[j + 3]);
  }
}

// ---------------------------------------------------------------------------
// 2) fp32 [K,N] -> bf16 [N,K] transpose-convert (weights)
// ---------------------------------------------------------------------------
__global__ void cvt_transpose_kernel(const float* __restrict__ in,
                                     bf16* __restrict__ out, int K, int N) {
  int idx = blockIdx.x * 256 + threadIdx.x;
  if (idx < K * N) {
    int kk = idx / N, n = idx - kk * N;
    out[(size_t)n * K + kk] = (bf16)in[idx];
  }
}

// ---------------------------------------------------------------------------
// 3/7) bf16 GEMM: out[M,N] = (A[M,K] @ Wt[N,K]^T + bias) * scale
//      Block tile 128x128, wave tile 32x64, K-step 32, DOUBLE-BUFFERED:
//      A tile via TDM tensor_load_to_lds (wave 0), B tile via async-to-LDS.
// ---------------------------------------------------------------------------
template <bool OUT_F32>
__global__ void gemm_kernel(const bf16* __restrict__ A,
                            const bf16* __restrict__ Wt,  // [N,K]
                            const float* __restrict__ bias,
                            void* __restrict__ outv,
                            int M, int K, int N, float scale) {
  __shared__ __align__(16) bf16 As[2][128][40];
  __shared__ __align__(16) bf16 Bt[2][128][40];

  const int tid   = threadIdx.x;
  const int lane  = tid & 31;
  const int w     = tid >> 5;          // 0..7
  const int wy    = w >> 1;            // 0..3 -> 32-row strip
  const int wx    = w & 1;             // 0..1 -> 64-col strip
  const int nbase = blockIdx.x * 128;
  const int mbase = blockIdx.y * 128;

  const int srow = tid >> 1;           // 0..127
  const int sc0  = (tid & 1) * 16;     // 0 or 16

  const bf16* Ab = A  + (size_t)mbase * K;
  const bf16* Bb = Wt + (size_t)(nbase + srow) * K + sc0;

  v8f acc[2][4] = {};

  const int nk = K / 32;
  // prologue: stage tile 0
  if (w == 0) tdm_load_tile_128x32(&As[0][0][0], Ab, (unsigned)K);
  async_copy16(&Bt[0][srow][sc0],     Bb);
  async_copy16(&Bt[0][srow][sc0 + 8], Bb + 8);

  for (int i = 0; i < nk; ++i) {
    const int cur  = i & 1;
    const int nxt  = cur ^ 1;
    const bool more = (i + 1) < nk;
    if (more) {
      const int kk = (i + 1) * 32;
      if (w == 0) tdm_load_tile_128x32(&As[nxt][0][0], Ab + kk, (unsigned)K);
      async_copy16(&Bt[nxt][srow][sc0],     Bb + kk);
      async_copy16(&Bt[nxt][srow][sc0 + 8], Bb + kk + 8);
      __builtin_prefetch(Bb + kk + 32, 0, 1);
    }
    if (w == 0) { if (more) tensor_wait<1>(); else tensor_wait<0>(); }
    if (more) async_wait<2>(); else async_wait<0>();
    __syncthreads();

    v16bf a[2], b[4];
#pragma unroll
    for (int ty = 0; ty < 2; ++ty)
      a[ty] = load_frag_a16x32(&As[cur][wy * 32 + ty * 16][0], 40);
#pragma unroll
    for (int tx = 0; tx < 4; ++tx)
      b[tx] = load_frag_b32x16(&Bt[cur][wx * 64 + tx * 16][0], 40);
#pragma unroll
    for (int ty = 0; ty < 2; ++ty)
#pragma unroll
      for (int tx = 0; tx < 4; ++tx)
        acc[ty][tx] = wmma_bf16(a[ty], b[tx], acc[ty][tx]);
    __syncthreads();
  }

  const int m0 = mbase + wy * 32;
  const int n0 = nbase + wx * 64;
  float* outf = (float*)outv;
  bf16*  outb = (bf16*)outv;

#pragma unroll
  for (int ty = 0; ty < 2; ++ty) {
#pragma unroll
    for (int tx = 0; tx < 4; ++tx) {
      const v8f cc = acc[ty][tx];
      const int ncol = n0 + tx * 16 + (lane & 15);
      const float bs = bias[ncol];
#pragma unroll
      for (int j = 0; j < 8; ++j) {
        const int mrow = m0 + ty * 16 + j + ((lane >> 4) << 3);
        const float val = (cc[j] + bs) * scale;
        if (OUT_F32) outf[(size_t)mrow * N + ncol] = val;
        else         outb[(size_t)mrow * N + ncol] = (bf16)val;
      }
    }
  }
}

// ---------------------------------------------------------------------------
// 4) Column softmax stats (softmax over QUERY axis)
// ---------------------------------------------------------------------------
__global__ void colstats_kernel(const bf16* __restrict__ q,
                                const bf16* __restrict__ k,
                                float* __restrict__ m_out,
                                float* __restrict__ z_out) {
  const int lane  = threadIdx.x & 31;
  const int w     = threadIdx.x >> 5;
  const int Tbase = blockIdx.x * 64 + w * 16;
  const int hq    = blockIdx.y;
  const int hkv   = hq >> 2;
  const int bz    = blockIdx.z;

  const bf16* kbase = k + (((size_t)bz * S + Tbase) * HKV + hkv) * HD;
  const v16bf kb0 = load_frag_b32x16(kbase, HKV * HD);
  const v16bf kb1 = load_frag_b32x16(kbase + 32, HKV * HD);

  float M = -INFINITY, Z = 0.f;

  for (int t0 = Tbase; t0 < S; t0 += 16) {
    const bf16* qbase = q + (((size_t)bz * S + t0) * HQ + hq) * HD;
    if (t0 + 16 < S) __builtin_prefetch(qbase + 16 * HQ * HD, 0, 1);
    v16bf qa0 = load_frag_a16x32(qbase, HQ * HD);
    v16bf qa1 = load_frag_a16x32(qbase + 32, HQ * HD);
    v8f c = {};
    c = wmma_bf16(qa0, kb0, c);
    c = wmma_bf16(qa1, kb1, c);

    if (t0 == Tbase) {  // diagonal tile: mask T > t
#pragma unroll
      for (int j = 0; j < 8; ++j) {
        const int t = t0 + j + ((lane >> 4) << 3);
        const int T = Tbase + (lane & 15);
        if (T > t) c[j] = NEGBIG;
      }
    }
    float tm = c[0];
#pragma unroll
    for (int j = 1; j < 8; ++j) tm = fmaxf(tm, c[j]);
    tm = fmaxf(tm, __shfl_xor(tm, 16));
    const float nM = fmaxf(M, tm);
    float ts = 0.f;
#pragma unroll
    for (int j = 0; j < 8; ++j) ts += __expf(c[j] - nM);
    ts += __shfl_xor(ts, 16);
    Z = Z * __expf(M - nM) + ts;
    M = nM;
  }

  if (lane < 16) {
    const size_t idx = ((size_t)(bz * HQ + hq)) * S + Tbase + lane;
    m_out[idx] = M;
    z_out[idx] = Z;
  }
}

// ---------------------------------------------------------------------------
// 5) V transpose: vb[b][T][hkv][dv] -> vt[b][hkv][dv][T] (K-major B operand)
// ---------------------------------------------------------------------------
__global__ void transpose_v_kernel(const bf16* __restrict__ vb,
                                   bf16* __restrict__ vt) {
  int idx = blockIdx.x * 256 + threadIdx.x;
  const int total = BB * S * HKV * HDV;
  if (idx < total) {
    int dv  = idx & (HDV - 1);
    int t1  = idx >> 6;          // /HDV
    int hkv = t1 & (HKV - 1);
    int t2  = t1 >> 3;           // /HKV
    int T   = t2 & (S - 1);
    int bz  = t2 >> 11;          // /S
    vt[(((size_t)bz * HKV + hkv) * HDV + dv) * S + T] = vb[idx];
  }
}

// ---------------------------------------------------------------------------
// 6) Attention output (barrier-free): per-wave 16 queries, V frags direct
//    from global vt, P relayout through per-wave LDS.
// ---------------------------------------------------------------------------
__global__ void attnout_kernel(const bf16* __restrict__ q,
                               const bf16* __restrict__ k,
                               const bf16* __restrict__ vt,
                               const float* __restrict__ ms,
                               const float* __restrict__ zs,
                               bf16* __restrict__ attn) {
  __shared__ __align__(16) bf16 Pb[4][16][32];  // per-wave P tile [t][T]

  const int tid  = threadIdx.x;
  const int lane = tid & 31;
  const int w    = tid >> 5;
  const int t0   = blockIdx.x * 64 + w * 16;
  const int hq   = blockIdx.y;
  const int hkv  = hq >> 2;
  const int bz   = blockIdx.z;

  const bf16* qbase = q + (((size_t)bz * S + t0) * HQ + hq) * HD;
  const v16bf qa0 = load_frag_a16x32(qbase, HQ * HD);
  const v16bf qa1 = load_frag_a16x32(qbase + 32, HQ * HD);

  const bf16* vtb = vt + ((size_t)(bz * HKV + hkv) * HDV) * S;

  v8f o[4] = {};

  const int kbmax = (t0 + 15) >> 5;  // per-wave, inclusive
  for (int kb = 0; kb <= kbmax; ++kb) {
    const int T0k = kb * 32;
#pragma unroll
    for (int tc = 0; tc < 2; ++tc) {
      const int Tc = T0k + tc * 16;
      const bf16* kb_p = k + (((size_t)bz * S + Tc) * HKV + hkv) * HD;
      if (kb < kbmax) __builtin_prefetch(kb_p + 32 * HKV * HD, 0, 1);
      v16bf b0 = load_frag_b32x16(kb_p, HKV * HD);
      v16bf b1 = load_frag_b32x16(kb_p + 32, HKV * HD);
      v8f s = {};
      s = wmma_bf16(qa0, b0, s);
      s = wmma_bf16(qa1, b1, s);

      const int Tcol = Tc + (lane & 15);
      const size_t sidx = ((size_t)(bz * HQ + hq)) * S + Tcol;
      const float mT = ms[sidx];
      const float rZ = 1.0f / zs[sidx];
#pragma unroll
      for (int j = 0; j < 8; ++j) {
        const int m = j + ((lane >> 4) << 3);
        const int t = t0 + m;
        const float p = (Tcol <= t) ? __expf(s[j] - mT) * rZ : 0.0f;
        Pb[w][m][tc * 16 + (lane & 15)] = (bf16)p;
      }
    }

    v16bf pa = load_frag_a16x32(&Pb[w][0][0], 32);
#pragma unroll
    for (int nv = 0; nv < 4; ++nv) {
      v16bf vb = load_frag_b32x16(vtb + (size_t)(nv * 16) * S + T0k, S);
      o[nv] = wmma_bf16(pa, vb, o[nv]);
    }
  }

#pragma unroll
  for (int nv = 0; nv < 4; ++nv) {
    const int dv = nv * 16 + (lane & 15);
#pragma unroll
    for (int j = 0; j < 8; ++j) {
      const int t = t0 + j + ((lane >> 4) << 3);
      attn[(((size_t)bz * S + t) * HQ + hq) * HDV + dv] = (bf16)o[nv][j];
    }
  }
}

// ---------------------------------------------------------------------------
// Host launcher
// ---------------------------------------------------------------------------
extern "C" void kernel_launch(void* const* d_in, const int* in_sizes, int n_in,
                              void* d_out, int out_size, void* d_ws, size_t ws_size,
                              hipStream_t stream) {
  const float* x    = (const float*)d_in[0];
  const float* ln_g = (const float*)d_in[1];
  const float* ln_b = (const float*)d_in[2];
  const float* Wq   = (const float*)d_in[3];
  const float* bq   = (const float*)d_in[4];
  const float* Wk   = (const float*)d_in[5];
  const float* bk   = (const float*)d_in[6];
  const float* Wv   = (const float*)d_in[7];
  const float* bv   = (const float*)d_in[8];
  const float* Wo   = (const float*)d_in[9];
  const float* bo   = (const float*)d_in[10];
  float* out        = (float*)d_out;

  const int M = BB * S;  // 4096
  char* ws = (char*)d_ws;
  size_t off = 0;
  auto alloc = [&](size_t bytes) -> void* {
    void* p = ws + off;
    off = (off + bytes + 255) & ~(size_t)255;
    return p;
  };
  bf16*  xn    = (bf16*)alloc((size_t)M * D * 2);
  bf16*  qb    = (bf16*)alloc((size_t)M * (HQ * HD) * 2);
  bf16*  kb    = (bf16*)alloc((size_t)M * (HKV * HD) * 2);
  bf16*  vb    = (bf16*)alloc((size_t)M * (HKV * HDV) * 2);
  bf16*  vt    = (bf16*)alloc((size_t)M * (HKV * HDV) * 2);
  bf16*  attnb = (bf16*)alloc((size_t)M * (HQ * HDV) * 2);
  bf16*  wqt   = (bf16*)alloc((size_t)D * (HQ * HD) * 2);   // [N,K]
  bf16*  wkt   = (bf16*)alloc((size_t)D * (HKV * HD) * 2);
  bf16*  wvt   = (bf16*)alloc((size_t)D * (HKV * HDV) * 2);
  bf16*  wot   = (bf16*)alloc((size_t)(HQ * HDV) * D * 2);
  float* m_st  = (float*)alloc((size_t)BB * HQ * S * 4);
  float* z_st  = (float*)alloc((size_t)BB * HQ * S * 4);

  ln_kernel<<<M, 256, 0, stream>>>(x, ln_g, ln_b, xn);

  auto cvt_t = [&](const float* src, bf16* dst, int K, int N) {
    int n = K * N;
    cvt_transpose_kernel<<<(n + 255) / 256, 256, 0, stream>>>(src, dst, K, N);
  };
  cvt_t(Wq, wqt, D, HQ * HD);
  cvt_t(Wk, wkt, D, HKV * HD);
  cvt_t(Wv, wvt, D, HKV * HDV);
  cvt_t(Wo, wot, HQ * HDV, D);

  gemm_kernel<false><<<dim3((HQ * HD) / 128, M / 128), 256, 0, stream>>>(
      xn, wqt, bq, qb, M, D, HQ * HD, QSCALE);
  gemm_kernel<false><<<dim3((HKV * HD) / 128, M / 128), 256, 0, stream>>>(
      xn, wkt, bk, kb, M, D, HKV * HD, 1.0f);
  gemm_kernel<false><<<dim3((HKV * HDV) / 128, M / 128), 256, 0, stream>>>(
      xn, wvt, bv, vb, M, D, HKV * HDV, 1.0f);

  colstats_kernel<<<dim3(S / 64, HQ, BB), 128, 0, stream>>>(qb, kb, m_st, z_st);

  {
    int n = BB * S * HKV * HDV;
    transpose_v_kernel<<<(n + 255) / 256, 256, 0, stream>>>(vb, vt);
  }

  attnout_kernel<<<dim3(S / 64, HQ, BB), 128, 0, stream>>>(qb, kb, vt, m_st,
                                                           z_st, attnb);

  gemm_kernel<true><<<dim3(D / 128, M / 128), 256, 0, stream>>>(
      attnb, wot, bo, out, M, HQ * HDV, D, 1.0f);
}
